// SigModuleParallel_cheng_v2_50165218018138
// MI455X (gfx1250) — matrix operand order repaired and checked by reading, not compiled, not run on hardware
//
#include <hip/hip_runtime.h>

// ---------------- problem constants (from reference) ----------------
#define NBATCH   16
#define CIN      64
#define JJ       25
#define TT       128
#define LPTS     9                 // points per stream (before basepoint)
#define SIC      8                 // conv out channels
#define NPOS     (NBATCH*JJ*TT*LPTS)   // 460800  conv positions
#define NSTREAM  (NBATCH*JJ*TT)        // 51200   signature streams
#define JT       (JJ*TT)               // 3200
#define JTL      (JJ*TT*LPTS)          // 28800
#define SIGCH    819               // 9 + 81 + 729
#define KPAD     832               // SIGCH padded to multiple of 32 for WMMA K
#define OUTD     64
#define EPSBN    1e-5f

typedef __attribute__((ext_vector_type(16))) _Float16 v16h;
typedef __attribute__((ext_vector_type(8)))  float    v8f;

__device__ __forceinline__ unsigned short f2h(float x) {
    union { _Float16 h; unsigned short u; } cv;
    cv.h = (_Float16)x;
    return cv.u;
}

// ---------------- K0: zero BN stat accumulators (replay-safe) ----------------
__global__ void k_zero_stats(float* __restrict__ stats) {
    if (threadIdx.x < 32) stats[threadIdx.x] = 0.0f;
}

// ---------------- K1: 1x1 conv + bias + ReLU + BN sum/sumsq ----------------
// x: [N,C,J,T,L]; one thread per (n,j,t,l) position, loop over C.
// y layout: [stream][l][sic] so the signature kernel streams it contiguously.
__global__ __launch_bounds__(256)
void k_conv_relu_stats(const float* __restrict__ x,
                       const float* __restrict__ cw,   // [SIC][CIN]
                       const float* __restrict__ cb,   // [SIC]
                       float* __restrict__ y,          // [NPOS][SIC]
                       float* __restrict__ stats) {    // sum[8], sumsq[8], ...
    __shared__ float wl[CIN * SIC];     // wl[c*8+o] (broadcast reads)
    __shared__ float ssum[SIC], ssq[SIC];
    const int tid = threadIdx.x;
    if (tid < CIN * SIC) {
        int c = tid >> 3, o = tid & 7;
        wl[tid] = cw[o * CIN + c];
    }
    if (tid < SIC) { ssum[tid] = 0.0f; ssq[tid] = 0.0f; }
    __syncthreads();

    const int p = blockIdx.x * 256 + tid;          // NPOS == 1800*256 exactly
    const int n = p / JTL;
    const int r = p % JTL;
    const float* xb = x + (size_t)n * (CIN * JTL) + r;

    float acc[SIC];
#pragma unroll
    for (int o = 0; o < SIC; ++o) acc[o] = cb[o];
    for (int c = 0; c < CIN; ++c) {
        float xv = xb[(size_t)c * JTL];            // coalesced across block
#pragma unroll
        for (int o = 0; o < SIC; ++o) acc[o] = fmaf(xv, wl[c * 8 + o], acc[o]);
    }
    float s8[SIC], q8[SIC];
#pragma unroll
    for (int o = 0; o < SIC; ++o) {
        float v = fmaxf(acc[o], 0.0f);             // ReLU before BN (per ref)
        acc[o] = v; s8[o] = v; q8[o] = v * v;
    }
    float* yr = y + (size_t)p * SIC;
#pragma unroll
    for (int o = 0; o < SIC; ++o) yr[o] = acc[o];

    // wave32 shuffle reduce, then LDS atomics, then one global atomic per block
#pragma unroll
    for (int o = 0; o < SIC; ++o) {
        float s = s8[o], q = q8[o];
        for (int d = 16; d > 0; d >>= 1) { s += __shfl_down(s, d); q += __shfl_down(q, d); }
        if ((tid & 31) == 0) { atomicAdd(&ssum[o], s); atomicAdd(&ssq[o], q); }
    }
    __syncthreads();
    if (tid < SIC) {
        atomicAdd(&stats[tid],     ssum[tid]);
        atomicAdd(&stats[8 + tid], ssq[tid]);
    }
}

// ---------------- K2: finalize BN scale/shift + pack lin_w to f16 ----------------
// w16p layout: k-pair interleaved, w16p[(k>>1)*128 + o*2 + (k&1)] = W[o][k]
// so a lane's B-fragment element pair (k, k+1) is one aligned dword.
__global__ __launch_bounds__(256)
void k_finalize(const float* __restrict__ gamma, const float* __restrict__ beta,
                const float* __restrict__ lw,     // [OUTD][SIGCH]
                float* __restrict__ stats,
                unsigned short* __restrict__ w16p) {
    const int tid = threadIdx.x;
    if (tid < SIC) {
        const float inv = 1.0f / (float)NPOS;
        float mean = stats[tid] * inv;
        float var  = stats[8 + tid] * inv - mean * mean;
        float sc   = gamma[tid] * rsqrtf(var + EPSBN);
        stats[16 + tid] = sc;
        stats[24 + tid] = beta[tid] - mean * sc;
    }
    for (int idx = tid; idx < KPAD * OUTD; idx += 256) {
        int k = idx >> 6, o = idx & 63;
        float v = (k < SIGCH) ? lw[o * SIGCH + k] : 0.0f;
        w16p[(k >> 1) * 128 + o * 2 + (k & 1)] = f2h(v);
    }
}

// ---------------- K3: fused depth-3 signature (to LDS, f16) + WMMA GEMM ----------------
// Block = 160 threads (5 waves) per 16-stream tile.
//   phase 1: 144 threads = (16 streams) x (9 rows i) run Chen's recursion;
//            thread (s,i) keeps s1[9], s2[i,:][9], s3[i,:,:][81] in VGPRs.
//   phase 2: 4 waves, one 16-wide N-tile each, K=832 in 26 WMMA steps.
__global__ __launch_bounds__(160)
void k_sig_gemm(const float* __restrict__ y,       // [NSTREAM][LPTS][SIC]
                const float* __restrict__ stats,   // scale @16, shift @24
                const unsigned short* __restrict__ w16p,
                const float* __restrict__ lb,      // [OUTD]
                float* __restrict__ out) {         // [NBATCH][OUTD][J][T]
    __shared__ __align__(16) unsigned short sig[16 * KPAD];  // 26 KB, f16 sig tile
    const int tid = threadIdx.x;
    const int st  = blockIdx.x;                    // stream tile (16 streams)

    if (tid < 144) {
        const int sl = tid & 15;                   // stream within tile
        const int i  = tid >> 4;                   // signature row index
        const int stream = st * 16 + sl;
        const float* yr = y + (size_t)stream * (LPTS * SIC);
        float sc[SIC], sh[SIC];
#pragma unroll
        for (int c = 0; c < SIC; ++c) { sc[c] = stats[16 + c]; sh[c] = stats[24 + c]; }

        float A[9], B[9], C[81], prev[9];
#pragma unroll
        for (int j = 0; j < 9; ++j) { A[j] = 0.f; B[j] = 0.f; prev[j] = 0.f; } // basepoint 0
#pragma unroll
        for (int j = 0; j < 81; ++j) C[j] = 0.f;

        for (int a = 0; a < 9; ++a) {              // 9 path increments
            float v[9];
            float cur0 = 0.125f * (float)a;        // time channel linspace(0,1,9)
            v[0] = cur0 - prev[0]; prev[0] = cur0;
#pragma unroll
            for (int c = 0; c < SIC; ++c) {
                float pv = fmaf(yr[a * SIC + c], sc[c], sh[c]);   // apply BN
                v[c + 1] = pv - prev[c + 1]; prev[c + 1] = pv;
            }
            // Chen step (row i view):
            //  s3[i,j,k] += s2old[i,j]*v[k] + (s1old[i] + v[i]/3) * (v[j]v[k]/2)
            //  s2[i,j]   += (s1old[i] + v[i]/2) * v[j]
            //  s1[j]     += v[j]
            const float Ai    = A[i];
            const float coefA = 0.5f * (Ai + v[i] * (1.0f / 3.0f));
            const float coefB = Ai + 0.5f * v[i];
#pragma unroll
            for (int j = 0; j < 9; ++j) {
                float cj = fmaf(coefA, v[j], B[j]);
#pragma unroll
                for (int k = 0; k < 9; ++k) C[j * 9 + k] = fmaf(cj, v[k], C[j * 9 + k]);
            }
#pragma unroll
            for (int j = 0; j < 9; ++j) B[j] = fmaf(coefB, v[j], B[j]);
#pragma unroll
            for (int j = 0; j < 9; ++j) A[j] += v[j];
        }
        // signatory layout: [s1(9) | s2 row-major(81) | s3 row-major(729) | pad]
        unsigned short* row = sig + sl * KPAD;
#pragma unroll
        for (int j = 0; j < 81; ++j) row[90 + i * 81 + j] = f2h(C[j]);
#pragma unroll
        for (int j = 0; j < 9; ++j)  row[9 + i * 9 + j]   = f2h(B[j]);
        if (i == 0) {
#pragma unroll
            for (int j = 0; j < 9; ++j) row[j] = f2h(A[j]);
            for (int k = SIGCH; k < KPAD; ++k) row[k] = 0;   // zero K padding
        }
    }
    __syncthreads();

    if (tid < 128) {
        const int wv   = tid >> 5;         // N tile: o in [wv*16, wv*16+16)
        const int lane = tid & 31;
        const int m    = lane & 15;        // A row (stream) / B,C,D column
        const int hi   = lane >> 4;
        const int o    = wv * 16 + m;
        const float bo = lb[o];
        v8f acc;
#pragma unroll
        for (int r = 0; r < 8; ++r) acc[r] = bo;   // bias folded into C

        const unsigned short* wrow = w16p + o * 2;
        for (int kt = 0; kt < 26; ++kt) {
            if (kt < 25) __builtin_prefetch((const void*)&wrow[(size_t)(kt + 1) * 2048], 0, 3);
            union { v16h v; unsigned int u[8]; } av, bv;
            // A fragment (contiguous k order, same as B): lanes 0-15 K=0..15,
            // lanes 16-31 K=16..31; pairs of consecutive K per dword.
#pragma unroll
            for (int r = 0; r < 8; ++r) {
                int kb = hi * 16 + 2 * r;
                av.u[r] = *(const unsigned int*)&sig[m * KPAD + kt * 32 + kb];
            }
            // B fragment (32x16): lanes 0-15 K=0..15, lanes 16-31 K=16..31;
            // w16p pair-interleave makes each element pair one dword.
#pragma unroll
            for (int r = 0; r < 8; ++r) {
                int kp = kt * 16 + hi * 8 + r;     // = (k tile-local base)/2
                bv.u[r] = *(const unsigned int*)&wrow[kp * 128];
            }
            acc = __builtin_amdgcn_wmma_f32_16x16x32_f16(
                false, av.v, false, bv.v, (short)0, acc, false, false);
        }
        // D layout: VGPR r -> M = r + hi*8, N = lane&15.  out[n][o][j][t]
        const int sbase = st * 16;
#pragma unroll
        for (int r = 0; r < 8; ++r) {
            int s = sbase + hi * 8 + r;
            out[(size_t)(s / JT) * (OUTD * JT) + (size_t)o * JT + (s % JT)] = acc[r];
        }
    }
}

// ---------------- launcher ----------------
extern "C" void kernel_launch(void* const* d_in, const int* in_sizes, int n_in,
                              void* d_out, int out_size, void* d_ws, size_t ws_size,
                              hipStream_t stream) {
    (void)in_sizes; (void)n_in; (void)out_size; (void)ws_size;
    const float* x     = (const float*)d_in[0];
    const float* cw    = (const float*)d_in[1];
    const float* cb    = (const float*)d_in[2];
    const float* gamma = (const float*)d_in[3];
    const float* beta  = (const float*)d_in[4];
    const float* lw    = (const float*)d_in[5];
    const float* lb    = (const float*)d_in[6];
    float* out = (float*)d_out;

    char* ws = (char*)d_ws;
    float*          y     = (float*)ws;                                   // 14,745,600 B
    float*          stats = (float*)(ws + (size_t)NPOS * SIC * 4);        // 128 B
    unsigned short* w16p  = (unsigned short*)(ws + (size_t)NPOS * SIC * 4 + 128); // 106,496 B

    k_zero_stats<<<1, 32, 0, stream>>>(stats);
    k_conv_relu_stats<<<NPOS / 256, 256, 0, stream>>>(x, cw, cb, y, stats);
    k_finalize<<<1, 256, 0, stream>>>(gamma, beta, lw, stats, w16p);
    k_sig_gemm<<<NSTREAM / 16, 160, 0, stream>>>(y, stats, w16p, lb, out);
}
